// Cross_Attention_8220567404811
// MI455X (gfx1250) — matrix operand order repaired and checked by reading, not compiled, not run on hardware
//
#include <hip/hip_runtime.h>
#include <hip/hip_bf16.h>

// ---------------------------------------------------------------------------
// Restormer-style channel ("transposed") cross-attention on MI455X (gfx1250).
// All 1x1 convs + attention GEMMs via v_wmma_f32_16x16x32_f16 (f32 accum),
// intermediates f16 (~1.2 GB HBM traffic -> bandwidth-bound at 23.3 TB/s).
// GEMM inputs are produced channel-contiguous ([b][p][c]) so BOTH WMMA
// fragments are contiguous per-lane b128 global loads: no LDS, no barriers
// in any GEMM hot loop (same codegen shape the compiler produced for attn).
// ---------------------------------------------------------------------------

#define BATCH 8
#define CDIM  256
#define HEADS 8
#define HDIM  32
#define HImg  128
#define WImg  128
#define HW    (HImg * WImg)       // 16384
#define NPIX  (BATCH * HW)        // 131072
#define KSPLIT 8                  // K-split factor for the q.k^T reduction

typedef __attribute__((ext_vector_type(16))) _Float16 v16h;
typedef __attribute__((ext_vector_type(8)))  float    v8f;

// 16-bit A-matrix 16x32 lane layout: lane = m + 16*g
//   halves sequential in K: K = 16g + h8 (same order as B)
__device__ __forceinline__ int kA(int h8, int g) {
  return h8 + 16 * g;
}
// B-matrix 32x16 lane layout used here: lane = n + 16*g, halves sequential in K (same order as A)
__device__ __forceinline__ int kB(int h8, int g) { return h8 + 16 * g; }

__device__ __forceinline__ float gelu_tanh(float v) {
  return 0.5f * v * (1.0f + tanhf(0.7978845608028654f * (v + 0.044715f * v * v * v)));
}

// ---------------------------------------------------------------------------
// One-shot f32 -> f16 weight conversion (weights then stay L2-resident).
// ---------------------------------------------------------------------------
__global__ void cvt_kernel(const float* __restrict__ in, _Float16* __restrict__ out) {
  int i = blockIdx.x * blockDim.x + threadIdx.x;
  out[i] = (_Float16)in[i];
}

// ---------------------------------------------------------------------------
// LayerNorm over channels (per pixel), f32 [b][c][hw] -> f16 TRANSPOSED
// [b][p][c] (channel-contiguous) so the GEMM B-fragments are b128 loads.
// ---------------------------------------------------------------------------
__global__ void ln_kernel(const float* __restrict__ x, const float* __restrict__ w,
                          const float* __restrict__ bb, _Float16* __restrict__ outT) {
  int p = blockIdx.x * blockDim.x + threadIdx.x;   // 0..NPIX-1 (exact)
  int bi = p >> 14;
  int pin = p & (HW - 1);
  const float* base = x + (size_t)bi * CDIM * HW + pin;
  float s = 0.f, ss = 0.f;
  for (int c = 0; c < CDIM; ++c) {
    float v = base[(size_t)c * HW];
    s += v; ss += v * v;
  }
  float mu = s * (1.0f / CDIM);
  float var = ss * (1.0f / CDIM) - mu * mu;
  float rstd = rsqrtf(var + 1e-5f);
  _Float16* ob = outT + ((size_t)bi * HW + pin) * CDIM;  // contiguous per pixel
  for (int c = 0; c < CDIM; ++c) {
    float v = (base[(size_t)c * HW] - mu) * rstd * w[c] + bb[c];
    ob[c] = (_Float16)v;
  }
}

// ---------------------------------------------------------------------------
// 1x1 conv as GEMM: out[b][m][p] = sum_c W[m][c] * actT[b][p][c] + bias[m]
// Block = 256 threads (8 waves), tile BM=128 x BN=32, K step 32, NO LDS.
// Per wave per K-step: A = 2 b128 loads (f16 weight row, L2-resident),
// B0/B1 = 2 b128 loads each (channel-contiguous pixel rows), 2 WMMAs.
// FINAL: f32 output + p_emb(f16) + hx(f32) fused epilogue (the ao conv).
// ---------------------------------------------------------------------------
template <bool FINAL>
__global__ void gemm_kernel(const _Float16* __restrict__ actT,  // [b][p][c]
                            const _Float16* __restrict__ wf,    // [m][c] f16
                            const float* __restrict__ bias,
                            _Float16* __restrict__ out16,       // [b][m][p]
                            float* __restrict__ outf,
                            const _Float16* __restrict__ pemb,  // [b][m][p]
                            const float* __restrict__ hx) {
  const int tid  = threadIdx.x;
  const int wv   = tid >> 5;
  const int lane = tid & 31;
  const int g    = lane >> 4;
  const int r    = lane & 15;

  const int bb = blockIdx.z;
  const int m0 = blockIdx.y * 128 + wv * 16;     // 16-row M tile per wave
  const int p0 = blockIdx.x * 32;                // two 16-col N tiles

  const _Float16* arow  = wf + (size_t)(m0 + r) * CDIM;
  const _Float16* b0row = actT + ((size_t)bb * HW + p0 + r) * CDIM;
  const _Float16* b1row = actT + ((size_t)bb * HW + p0 + 16 + r) * CDIM;

  v8f acc0 = {0.f, 0.f, 0.f, 0.f, 0.f, 0.f, 0.f, 0.f};
  v8f acc1 = {0.f, 0.f, 0.f, 0.f, 0.f, 0.f, 0.f, 0.f};

  for (int k0 = 0; k0 < CDIM; k0 += 32) {
    v16h av, bv0, bv1;
#pragma unroll
    for (int h8 = 0; h8 < 16; ++h8) av[h8] = arow[k0 + kA(h8, g)];
#pragma unroll
    for (int h8 = 0; h8 < 16; ++h8) {
      int kk = k0 + kB(h8, g);
      bv0[h8] = b0row[kk];
      bv1[h8] = b1row[kk];
    }
    acc0 = __builtin_amdgcn_wmma_f32_16x16x32_f16(false, av, false, bv0,
                                                  (short)0, acc0, false, false);
    acc1 = __builtin_amdgcn_wmma_f32_16x16x32_f16(false, av, false, bv1,
                                                  (short)0, acc1, false, false);
  }

  // epilogue: D[m][n], m = vgpr + 8g, n = r
#pragma unroll
  for (int vv = 0; vv < 8; ++vv) {
    int m = m0 + vv + 8 * g;
    float bsv = bias[m];
    size_t rowb = ((size_t)bb * gridDim.y * 128 + m) * HW;
    int n0 = p0 + r;
    float v0 = acc0[vv] + bsv;
    float v1 = acc1[vv] + bsv;
    if (FINAL) {
      outf[rowb + n0]      = v0 + (float)pemb[rowb + n0]      + hx[rowb + n0];
      outf[rowb + n0 + 16] = v1 + (float)pemb[rowb + n0 + 16] + hx[rowb + n0 + 16];
    } else {
      out16[rowb + n0]      = (_Float16)v0;
      out16[rowb + n0 + 16] = (_Float16)v1;
    }
  }
}

// ---------------------------------------------------------------------------
// Depthwise 3x3 'SAME' (cross-correlation, per XLA), optional bias / GELU.
// in channels addressed with (CHT, CO) so we can slice the v-half of kv.
// ---------------------------------------------------------------------------
template <bool GELU, bool BIAS>
__global__ void dw3_kernel(const _Float16* __restrict__ in, int CHT, int CO,
                           const float* __restrict__ w,
                           const float* __restrict__ bias,
                           _Float16* __restrict__ out) {
  int p  = blockIdx.x * blockDim.x + threadIdx.x;  // 0..HW-1
  int y  = p >> 7;
  int xc = p & 127;
  int ch = blockIdx.y;
  int bt = blockIdx.z;
  const _Float16* ib = in + ((size_t)bt * CHT + CO + ch) * HW;
  const float* wp = w + ch * 9;
  float acc = BIAS ? bias[ch] : 0.f;
#pragma unroll
  for (int dy = -1; dy <= 1; ++dy) {
    int yy = y + dy;
    if (yy < 0 || yy >= HImg) continue;
#pragma unroll
    for (int dx = -1; dx <= 1; ++dx) {
      int xx = xc + dx;
      if (xx < 0 || xx >= WImg) continue;
      acc += (float)ib[yy * WImg + xx] * wp[(dy + 1) * 3 + (dx + 1)];
    }
  }
  if (GELU) acc = gelu_tanh(acc);
  out[((size_t)bt * gridDim.y + ch) * HW + p] = (_Float16)acc;
}

// ---------------------------------------------------------------------------
// Per-(b,c) inverse L2 norm over the 16384 spatial elements.
// ---------------------------------------------------------------------------
__global__ void rnorm_kernel(const _Float16* __restrict__ in, int CHT, int CO,
                             float* __restrict__ rinv) {
  __shared__ float red[256];
  int row = blockIdx.x;              // b*256 + c
  int bt = row >> 8, c = row & 255;
  const _Float16* rp = in + ((size_t)bt * CHT + CO + c) * HW;
  float s = 0.f;
  for (int i = threadIdx.x; i < HW; i += 256) {
    float v = (float)rp[i];
    s += v * v;
  }
  red[threadIdx.x] = s;
  __syncthreads();
  for (int st = 128; st > 0; st >>= 1) {
    if (threadIdx.x < st) red[threadIdx.x] += red[threadIdx.x + st];
    __syncthreads();
  }
  if (threadIdx.x == 0) rinv[row] = 1.0f / fmaxf(sqrtf(red[0]), 1e-12f);
}

// ---------------------------------------------------------------------------
// Raw partial attn logits: part[ks][b,h][c][d] = sum_{p in slice} q[c,p]*k[d,p]
// grid (64, KSPLIT); 4 waves/block; pure b128 loads + WMMA + WGP-scope
// prefetch. Deterministic: each block owns its partial slot (no atomics).
// ---------------------------------------------------------------------------
__global__ void attn_kernel(const _Float16* __restrict__ qd,
                            const _Float16* __restrict__ kd,
                            float* __restrict__ attnp) {
  const int bh = blockIdx.x;
  const int ks = blockIdx.y;
  const int bt = bh >> 3, h = bh & 7;
  const int tid = threadIdx.x, wv = tid >> 5, lane = tid & 31;
  const int g = lane >> 4, r = lane & 15;
  const int ct = wv >> 1, dt = wv & 1;
  const int pbeg = ks * (HW / KSPLIT);
  const int pend = pbeg + (HW / KSPLIT);

  const _Float16* qrow = qd + ((size_t)bt * CDIM + h * HDIM + ct * 16 + r) * HW;
  const _Float16* krow = kd + ((size_t)bt * (2 * CDIM) + h * HDIM + dt * 16 + r) * HW;

  v8f acc = {0.f, 0.f, 0.f, 0.f, 0.f, 0.f, 0.f, 0.f};
  for (int p0 = pbeg; p0 < pend; p0 += 32) {
    if (p0 + 128 < pend) {   // WGP-scope streaming prefetch (all cache levels)
      __builtin_prefetch(qrow + p0 + 128, 0, 3);
      __builtin_prefetch(krow + p0 + 128, 0, 3);
    }
    v16h av, bv;
#pragma unroll
    for (int h8 = 0; h8 < 16; ++h8) {
      av[h8] = qrow[p0 + kA(h8, g)];
      bv[h8] = krow[p0 + kB(h8, g)];
    }
    acc = __builtin_amdgcn_wmma_f32_16x16x32_f16(false, av, false, bv,
                                                 (short)0, acc, false, false);
  }
#pragma unroll
  for (int vv = 0; vv < 8; ++vv) {
    int ci = ct * 16 + vv + 8 * g;
    int di = dt * 16 + r;
    attnp[((size_t)ks * 64 + bh) * 1024 + ci * 32 + di] = acc[vv];
  }
}

// ---------------------------------------------------------------------------
// Sum the KSPLIT partials, apply rq*rk*temperature, row softmax over d,
// emit f16 attention. One block per (b,h); 32 lanes per row (shuffle reduce).
// ---------------------------------------------------------------------------
__global__ void softmax_kernel(const float* __restrict__ attnp,
                               const float* __restrict__ rq,
                               const float* __restrict__ rk,
                               const float* __restrict__ temp,
                               _Float16* __restrict__ a16) {
  const int bh = blockIdx.x;
  const int bt = bh >> 3, h = bh & 7;
  int tid = threadIdx.x;
  int row = tid >> 5, col = tid & 31;
  float s = 0.f;
#pragma unroll
  for (int ks = 0; ks < KSPLIT; ++ks)
    s += attnp[((size_t)ks * 64 + bh) * 1024 + row * 32 + col];
  float v = s * rq[bt * CDIM + h * HDIM + row]
              * rk[bt * CDIM + h * HDIM + col] * temp[h];
  float m = v;
  for (int st = 16; st > 0; st >>= 1) m = fmaxf(m, __shfl_xor(m, st, 32));
  float e = __expf(v - m);
  float sum = e;
  for (int st = 16; st > 0; st >>= 1) sum += __shfl_xor(sum, st, 32);
  a16[(size_t)bh * 1024 + row * 32 + col] = (_Float16)(e / sum);
}

// ---------------------------------------------------------------------------
// out[b][p][h*32+c] (TRANSPOSED, channel-contiguous) = sum_d attn[c,d]*v[d,p]
// K = 32 -> one WMMA per tile; v tile staged transposed in LDS.
// Transposed output makes the epilogue one contiguous b128 store per lane
// and feeds the LDS-free ao GEMM directly.
// ---------------------------------------------------------------------------
__global__ void attn_out_kernel(const _Float16* __restrict__ a16,
                                const _Float16* __restrict__ kvd,
                                _Float16* __restrict__ aoutT) {
  __shared__ _Float16 vt[64][40];    // [p_local][d], padded
  const int bh = blockIdx.y;
  const int bt = bh >> 3, h = bh & 7;
  const int p0 = blockIdx.x * 64;
  const int tid = threadIdx.x, wv = tid >> 5, lane = tid & 31;
  const int g = lane >> 4, r = lane & 15;

  // stage v tile transposed: vt[p][d] = v[d][p0+p]
  {
    int d = tid >> 3;
    int px = (tid & 7) * 8;
    const _Float16* src =
        kvd + ((size_t)bt * (2 * CDIM) + CDIM + h * HDIM + d) * HW + p0 + px;
#pragma unroll
    for (int j = 0; j < 8; ++j) vt[px + j][d] = src[j];
  }
  __syncthreads();

  const int ct = wv >> 2;            // 0..1 (channel half of head)
  const int pt = wv & 3;             // 0..3 (16-pixel tile)

  v16h av, bv;
  const _Float16* arow = a16 + (size_t)bh * 1024 + (ct * 16 + r) * 32;
#pragma unroll
  for (int h8 = 0; h8 < 16; ++h8) {
    av[h8] = arow[kA(h8, g)];
    bv[h8] = vt[pt * 16 + r][kB(h8, g)];
  }
  v8f acc = {0.f, 0.f, 0.f, 0.f, 0.f, 0.f, 0.f, 0.f};
  acc = __builtin_amdgcn_wmma_f32_16x16x32_f16(false, av, false, bv,
                                               (short)0, acc, false, false);
  // lane's 8 D values are channel-contiguous in the transposed layout
  int n = p0 + pt * 16 + r;
  _Float16* dst = aoutT + ((size_t)bt * HW + n) * CDIM + h * HDIM + ct * 16 + 8 * g;
#pragma unroll
  for (int vv = 0; vv < 8; ++vv) dst[vv] = (_Float16)acc[vv];
}

// ---------------------------------------------------------------------------
// Host-side launch sequence (all on `stream`; workspace carved from d_ws).
// ---------------------------------------------------------------------------
extern "C" void kernel_launch(void* const* d_in, const int* in_sizes, int n_in,
                              void* d_out, int out_size, void* d_ws, size_t ws_size,
                              hipStream_t stream) {
  const float* x      = (const float*)d_in[0];
  const float* hx     = (const float*)d_in[1];
  const float* ln1_w  = (const float*)d_in[2];
  const float* ln1_b  = (const float*)d_in[3];
  const float* ln2_w  = (const float*)d_in[4];
  const float* ln2_b  = (const float*)d_in[5];
  const float* q_w    = (const float*)d_in[6];
  const float* q_b    = (const float*)d_in[7];
  const float* q_dw_w = (const float*)d_in[8];
  const float* q_dw_b = (const float*)d_in[9];
  const float* kv_w   = (const float*)d_in[10];
  const float* kv_b   = (const float*)d_in[11];
  const float* kv_dw_w= (const float*)d_in[12];
  const float* kv_dw_b= (const float*)d_in[13];
  const float* ao_w   = (const float*)d_in[14];
  const float* ao_b   = (const float*)d_in[15];
  const float* pe1_w  = (const float*)d_in[16];
  const float* pe2_w  = (const float*)d_in[17];
  const float* temp   = (const float*)d_in[18];

  char* ws = (char*)d_ws;
  const size_t MB = 1ull << 20;
  _Float16* W0 = (_Float16*)(ws);              // 64 MB: lnT(x)/lnT(hx)/attn_outT
  _Float16* W1 = (_Float16*)(ws + 64 * MB);    // 64 MB: q / gelu(pe1)
  _Float16* W2 = (_Float16*)(ws + 128 * MB);   // 128 MB: kv / p_emb
  _Float16* W3 = (_Float16*)(ws + 256 * MB);   // 64 MB: q after dwconv
  _Float16* W4 = (_Float16*)(ws + 320 * MB);   // 128 MB: kv after dwconv
  char* sm = ws + 448 * MB;
  float*    RQ  = (float*)sm;                          sm += BATCH * CDIM * 4;
  float*    RK  = (float*)sm;                          sm += BATCH * CDIM * 4;
  float*    ATP = (float*)sm;                          sm += (size_t)KSPLIT * 64 * 1024 * 4;
  _Float16* A16 = (_Float16*)sm;                       sm += 64 * 1024 * 2;
  _Float16* WQ  = (_Float16*)sm;                       sm += (size_t)CDIM * CDIM * 2;
  _Float16* WKV = (_Float16*)sm;                       sm += (size_t)2 * CDIM * CDIM * 2;
  _Float16* WAO = (_Float16*)sm;

  // 0) one-shot weight conversion to f16 (L2-resident afterwards)
  cvt_kernel<<<(CDIM * CDIM) / 256, 256, 0, stream>>>(q_w, WQ);
  cvt_kernel<<<(2 * CDIM * CDIM) / 256, 256, 0, stream>>>(kv_w, WKV);
  cvt_kernel<<<(CDIM * CDIM) / 256, 256, 0, stream>>>(ao_w, WAO);

  // 1) LN(x) -> W0 (transposed) ; q = W @ LN(x) + b -> W1 [b][c][p]
  ln_kernel<<<NPIX / 256, 256, 0, stream>>>(x, ln1_w, ln1_b, W0);
  gemm_kernel<false><<<dim3(HW / 32, CDIM / 128, BATCH), 256, 0, stream>>>(
      W0, WQ, q_b, W1, nullptr, nullptr, nullptr);

  // 2) LN(hx) -> W0 (reuse) ; kv -> W2
  ln_kernel<<<NPIX / 256, 256, 0, stream>>>(hx, ln2_w, ln2_b, W0);
  gemm_kernel<false><<<dim3(HW / 32, (2 * CDIM) / 128, BATCH), 256, 0, stream>>>(
      W0, WKV, kv_b, W2, nullptr, nullptr, nullptr);

  // 3) depthwise convs
  dw3_kernel<false, true><<<dim3(HW / 256, CDIM, BATCH), 256, 0, stream>>>(
      W1, CDIM, 0, q_dw_w, q_dw_b, W3);
  dw3_kernel<false, true><<<dim3(HW / 256, 2 * CDIM, BATCH), 256, 0, stream>>>(
      W2, 2 * CDIM, 0, kv_dw_w, kv_dw_b, W4);

  // 4) positional branch: gelu(dw(v, pe1)) -> W1 ; dw(., pe2) -> W2 (p_emb)
  dw3_kernel<true, false><<<dim3(HW / 256, CDIM, BATCH), 256, 0, stream>>>(
      W4, 2 * CDIM, CDIM, pe1_w, nullptr, W1);
  dw3_kernel<false, false><<<dim3(HW / 256, CDIM, BATCH), 256, 0, stream>>>(
      W1, CDIM, 0, pe2_w, nullptr, W2);

  // 5) q/k inverse norms
  rnorm_kernel<<<BATCH * CDIM, 256, 0, stream>>>(W3, CDIM, 0, RQ);
  rnorm_kernel<<<BATCH * CDIM, 256, 0, stream>>>(W4, 2 * CDIM, 0, RK);

  // 6) raw q.k^T partials (K split 8-ways), then scale+softmax
  attn_kernel<<<dim3(BATCH * HEADS, KSPLIT), 128, 0, stream>>>(W3, W4, ATP);
  softmax_kernel<<<BATCH * HEADS, 1024, 0, stream>>>(ATP, RQ, RK, temp, A16);

  // 7) out = attn @ v -> W0 (transposed [b][p][c])
  attn_out_kernel<<<dim3(HW / 64, BATCH * HEADS), 256, 0, stream>>>(A16, W4, W0);

  // 8) final: ao conv + p_emb + hx -> d_out (f32)
  gemm_kernel<true><<<dim3(HW / 32, CDIM / 128, BATCH), 256, 0, stream>>>(
      W0, WAO, ao_b, nullptr, (float*)d_out, W2, hx);
}
